// Gemma3AttentionLearn_76725295776108
// MI455X (gfx1250) — matrix operand order repaired and checked by reading, not compiled, not run on hardware
//
#include <hip/hip_runtime.h>
#include <math.h>

typedef __attribute__((ext_vector_type(16))) __bf16 bf16x16;
typedef __attribute__((ext_vector_type(8)))  float  f32x8;

#define DEVINL __device__ __forceinline__

// ---- constants of the problem ----
static constexpr int Bc  = 2;
static constexpr int Sc  = 2048;
static constexpr int HIDc = 2560;
static constexpr int Hc  = 8;
static constexpr int KVc = 4;
static constexpr int HDc = 256;
static constexpr float SCALINGc = 0.0625f;   // 256^-0.5
static constexpr float EPSc = 1e-6f;
static constexpr float NEGc = -1e9f;

// float -> bf16 (round to nearest even)
DEVINL __bf16 f2bf(float f) {
  unsigned u = __builtin_bit_cast(unsigned, f);
  u += 0x7FFFu + ((u >> 16) & 1u);
  unsigned short h = (unsigned short)(u >> 16);
  return __builtin_bit_cast(__bf16, h);
}

DEVINL f32x8 wmma_bf16(const bf16x16& a, const bf16x16& b, const f32x8& c) {
  return __builtin_amdgcn_wmma_f32_16x16x32_bf16(false, a, false, b, (short)0, c,
                                                 false, false);
}

// =====================================================================
// f32 -> bf16 bulk convert, 8 elements/thread (vectorizes to b128 ld/st)
// n must be a multiple of 2048
// =====================================================================
__global__ void __launch_bounds__(256)
convert_bf16(const float* __restrict__ in, __bf16* __restrict__ out) {
  const size_t base = ((size_t)blockIdx.x * 256 + threadIdx.x) * 8;
  const float* ip = in + base;
  __bf16* op = out + base;
#pragma unroll
  for (int j = 0; j < 8; ++j) op[j] = f2bf(ip[j]);
}

// =====================================================================
// GEMM: Y[M,N] = A[M,K] @ W[N,K]^T   (bf16 in, f32 WMMA accum/out)
// block = 256 threads = 8 waves; wave tile = 64(M) x 16(N); block = 64 x 128
// grid = (M/64, N/128).  Main loop is pure b128-load + WMMA.
// =====================================================================
__global__ void __launch_bounds__(256)
gemm_nt(const __bf16* __restrict__ A, const __bf16* __restrict__ W,
        float* __restrict__ Y, int M, int N, int K) {
  const int lane  = threadIdx.x & 31;
  const int wave  = threadIdx.x >> 5;
  const int mBase = blockIdx.x * 64;
  const int nTile = blockIdx.y * 128 + wave * 16;
  const int mloc  = lane & 15;
  const int hi    = lane >> 4;
  const int kb0   = hi * 8;    // unused; A fragment also uses kb2
  const int kb2   = hi * 16;   // B-fragment K chunk base

  f32x8 acc[4] = {};

  for (int kb = 0; kb < K; kb += 32) {
    // B fragment: lane n = mloc, 16 consecutive K values at kb+kb2
    bf16x16 bfrag;
    {
      const __bf16* bp = W + (size_t)(nTile + mloc) * K + kb + kb2;
#pragma unroll
      for (int j = 0; j < 16; ++j) bfrag[j] = bp[j];
    }
#pragma unroll
    for (int msub = 0; msub < 4; ++msub) {
      const __bf16* ap = A + (size_t)(mBase + msub * 16 + mloc) * K + kb;
      bf16x16 afrag;
#pragma unroll
      for (int j = 0; j < 8; ++j) {
        afrag[j]     = ap[kb2 + j];
        afrag[8 + j] = ap[kb2 + 8 + j];
      }
      acc[msub] = wmma_bf16(afrag, bfrag, acc[msub]);
    }
  }
#pragma unroll
  for (int msub = 0; msub < 4; ++msub) {
#pragma unroll
    for (int r = 0; r < 8; ++r) {
      int row = mBase + msub * 16 + r + hi * 8;
      Y[(size_t)row * N + nTile + mloc] = acc[msub][r];
    }
  }
}

// =====================================================================
// QK RMSNorm + RoPE.  One wave per (b,h,s) row of 256 dims.
// in : t_raw [(b*S+s)*Hn*HD + h*HD + d]   (f32, GEMM output layout)
// out: [B,Hn,S,HD] bf16, scaled by outScale (softmax scale folded into Q)
// grid = B*Hn*S/8 blocks of 256
// =====================================================================
__global__ void __launch_bounds__(256)
qknorm_rope(const float* __restrict__ t_raw, const float* __restrict__ normw,
            const float* __restrict__ cosT, const float* __restrict__ sinT,
            __bf16* __restrict__ out, int Hn, float outScale) {
  const int lane = threadIdx.x & 31;
  const int row  = blockIdx.x * 8 + (threadIdx.x >> 5);   // over b,h,s
  const int s = row % Sc;
  const int h = (row / Sc) % Hn;
  const int b = row / (Sc * Hn);
  const int d0 = lane * 8;

  const float* in = t_raw + ((size_t)(b * Sc + s) * Hn + h) * HDc + d0;
  float v[8];
  float ss = 0.f;
#pragma unroll
  for (int j = 0; j < 8; ++j) { v[j] = in[j]; ss += v[j] * v[j]; }
#pragma unroll
  for (int d = 1; d < 32; d <<= 1) ss += __shfl_xor(ss, d);
  const float inv = rsqrtf(ss * (1.0f / HDc) + EPSc);

  float nw[8];
#pragma unroll
  for (int j = 0; j < 8; ++j) nw[j] = v[j] * inv * (1.0f + normw[d0 + j]);

  __bf16* op = out + (((size_t)(b * Hn + h) * Sc) + s) * HDc + d0;
  const float* cp = cosT + (size_t)s * HDc + d0;
  const float* sp = sinT + (size_t)s * HDc + d0;
#pragma unroll
  for (int j = 0; j < 8; ++j) {
    float partner = __shfl_xor(nw[j], 16);              // dim +/- 128
    float rot = (d0 < 128) ? -partner : partner;        // rotate_half
    float o = (nw[j] * cp[j] + rot * sp[j]) * outScale;
    op[j] = f2bf(o);
  }
}

// =====================================================================
// V transpose: [B,S,KV,HD] f32 -> [B,KV,HD,S] bf16 (dim-major for P@V)
// grid = (S/256, B*KV*HD); cheap 2D index math, coalesced along s
// =====================================================================
__global__ void __launch_bounds__(256)
v_transpose(const float* __restrict__ v_raw, __bf16* __restrict__ v_t) {
  const int s   = blockIdx.x * 256 + threadIdx.x;
  const int row = blockIdx.y;                 // (b*KV + kv)*HD + d
  const int d   = row % HDc;
  const int kv  = (row / HDc) % KVc;
  const int b   = row / (HDc * KVc);
  v_t[(size_t)row * Sc + s] =
      f2bf(v_raw[((size_t)(b * Sc + s) * KVc + kv) * HDc + d]);
}

// =====================================================================
// Flash attention (causal, GQA).  One wave = one (b,h, 16-query tile).
// q [B,H,S,HD] bf16 (pre-scaled), k [B,KV,S,HD] bf16, vt [B,KV,HD,S] bf16
// o [B*S, H*HD] bf16
// grid = B*H*(S/16)/8 blocks of 256
// =====================================================================
__global__ void __launch_bounds__(256)
attn(const __bf16* __restrict__ qg, const __bf16* __restrict__ kg,
     const __bf16* __restrict__ vtg, __bf16* __restrict__ og) {
  const int lane = threadIdx.x & 31;
  const int wave = threadIdx.x >> 5;
  const int gw   = blockIdx.x * 8 + wave;
  const int qt   = gw % (Sc / 16);
  const int h    = (gw / (Sc / 16)) % Hc;
  const int b    = gw / ((Sc / 16) * Hc);
  const int kv   = h / (Hc / KVc);
  const int qBase = qt * 16;
  const int mloc = lane & 15;
  const int hi   = lane >> 4;
  const int kb0  = hi * 8;
  const int kb2  = hi * 16;

  __shared__ __bf16 Pbuf[8][16 * 32];
  __bf16* pb = &Pbuf[wave][0];

  // Preload Q fragments for all 8 K(=dim)-blocks of 32
  bf16x16 qfrag[8];
  {
    const __bf16* qrow = qg + (((size_t)(b * Hc + h)) * Sc + qBase + mloc) * HDc;
#pragma unroll
    for (int jb = 0; jb < 8; ++jb) {
#pragma unroll
      for (int j = 0; j < 8; ++j) {
        qfrag[jb][j]     = qrow[jb * 32 + kb2 + j];
        qfrag[jb][8 + j] = qrow[jb * 32 + kb2 + 8 + j];
      }
    }
  }

  f32x8 acc[16] = {};
  float mstat[8], lstat[8];
#pragma unroll
  for (int r = 0; r < 8; ++r) { mstat[r] = -__builtin_inff(); lstat[r] = 0.f; }

  const __bf16* kbp = kg  + ((size_t)(b * KVc + kv)) * Sc * HDc;
  const __bf16* vbp = vtg + ((size_t)(b * KVc + kv)) * HDc * Sc;

  for (int kc = 0; kc < qBase + 16; kc += 32) {
    // ---- scores: two 16-key tiles ----
    f32x8 s0 = {}, s1 = {};
#pragma unroll
    for (int jb = 0; jb < 8; ++jb) {
      bf16x16 kf0, kf1;
      const __bf16* kp0 = kbp + (size_t)(kc + mloc) * HDc + jb * 32 + kb2;
      const __bf16* kp1 = kbp + (size_t)(kc + 16 + mloc) * HDc + jb * 32 + kb2;
#pragma unroll
      for (int j = 0; j < 16; ++j) { kf0[j] = kp0[j]; kf1[j] = kp1[j]; }
      s0 = wmma_bf16(qfrag[jb], kf0, s0);
      s1 = wmma_bf16(qfrag[jb], kf1, s1);
    }

    // ---- causal mask + online softmax (row r+8*hi lives on one half-wave) --
#pragma unroll
    for (int r = 0; r < 8; ++r) {
      const int qq = qBase + r + hi * 8;
      float a0 = s0[r] + (((kc + mloc) > qq) ? NEGc : 0.f);
      float a1 = s1[r] + (((kc + 16 + mloc) > qq) ? NEGc : 0.f);
      float mx = fmaxf(a0, a1);
#pragma unroll
      for (int d = 1; d < 16; d <<= 1) mx = fmaxf(mx, __shfl_xor(mx, d));
      const float mnew  = fmaxf(mstat[r], mx);
      const float scale = __expf(mstat[r] - mnew);
      const float p0 = __expf(a0 - mnew);
      const float p1 = __expf(a1 - mnew);
      float rs = p0 + p1;
#pragma unroll
      for (int d = 1; d < 16; d <<= 1) rs += __shfl_xor(rs, d);
      lstat[r] = lstat[r] * scale + rs;
      mstat[r] = mnew;
#pragma unroll
      for (int t = 0; t < 16; ++t) acc[t][r] *= scale;
      const int rowoff = (r + hi * 8) * 32;
      pb[rowoff + mloc]      = f2bf(p0);
      pb[rowoff + 16 + mloc] = f2bf(p1);
    }

    // make the LDS-staged P visible to the A-fragment reads (same wave)
    asm volatile("s_wait_dscnt 0" ::: "memory");

    // ---- P (A-frag from LDS) @ V ----
    bf16x16 pf;
#pragma unroll
    for (int j = 0; j < 8; ++j) {
      pf[j]     = pb[mloc * 32 + kb2 + j];
      pf[8 + j] = pb[mloc * 32 + kb2 + 8 + j];
    }
#pragma unroll
    for (int t = 0; t < 16; ++t) {
      bf16x16 vf;
      const __bf16* vp = vbp + (size_t)(t * 16 + mloc) * Sc + kc + kb2;
#pragma unroll
      for (int j = 0; j < 16; ++j) vf[j] = vp[j];
      acc[t] = wmma_bf16(pf, vf, acc[t]);
    }
  }

  // ---- normalize + store O (bf16, [B*S, H*HD]) ----
#pragma unroll
  for (int r = 0; r < 8; ++r) {
    const float invl = 1.0f / lstat[r];
    const size_t row = (size_t)(b * Sc + qBase + r + hi * 8);
#pragma unroll
    for (int t = 0; t < 16; ++t) {
      og[row * (Hc * HDc) + h * HDc + t * 16 + mloc] = f2bf(acc[t][r] * invl);
    }
  }
}

// =====================================================================
// Host-side launcher
// =====================================================================
extern "C" void kernel_launch(void* const* d_in, const int* in_sizes, int n_in,
                              void* d_out, int out_size, void* d_ws, size_t ws_size,
                              hipStream_t stream) {
  const float* x   = (const float*)d_in[0];
  const float* wq  = (const float*)d_in[1];
  const float* wk  = (const float*)d_in[2];
  const float* wv  = (const float*)d_in[3];
  const float* wo  = (const float*)d_in[4];
  const float* qnw = (const float*)d_in[5];
  const float* knw = (const float*)d_in[6];
  const float* cosT = (const float*)d_in[7];
  const float* sinT = (const float*)d_in[8];
  // d_in[9] = mask: causality is applied analytically in the kernel.
  float* out = (float*)d_out;

  const int M  = Bc * Sc;          // 4096
  const int NQ = Hc * HDc;         // 2048
  const int NK = KVc * HDc;        // 1024

  char* ws = (char*)d_ws;
  float*  q_raw = (float*)(ws);                               // 4096*2048*4
  float*  k_raw = (float*)(ws + 33554432);                    // 4096*1024*4
  float*  v_raw = (float*)(ws + 50331648);                    // 4096*1024*4
  __bf16* q_bf  = (__bf16*)(ws + 67108864);                   // 4096*2048*2
  __bf16* k_bf  = (__bf16*)(ws + 83886080);                   // 4096*1024*2
  __bf16* v_t   = (__bf16*)(ws + 92274688);                   // 4096*1024*2
  __bf16* o_bf  = (__bf16*)(ws + 100663296);                  // 4096*2048*2
  __bf16* x_bf  = (__bf16*)(ws + 117440512);                  // 4096*2560*2
  __bf16* wq_bf = (__bf16*)(ws + 138412032);                  // 2048*2560*2
  __bf16* wk_bf = (__bf16*)(ws + 148897792);                  // 1024*2560*2
  __bf16* wv_bf = (__bf16*)(ws + 154140672);                  // 1024*2560*2
  __bf16* wo_bf = (__bf16*)(ws + 159383552);                  // 2560*2048*2

  // One-time bf16 conversions (memory-bound, trivial vs. the GEMMs)
  convert_bf16<<<(M * HIDc) / 2048, 256, 0, stream>>>(x, x_bf);
  convert_bf16<<<(NQ * HIDc) / 2048, 256, 0, stream>>>(wq, wq_bf);
  convert_bf16<<<(NK * HIDc) / 2048, 256, 0, stream>>>(wk, wk_bf);
  convert_bf16<<<(NK * HIDc) / 2048, 256, 0, stream>>>(wv, wv_bf);
  convert_bf16<<<(HIDc * NQ) / 2048, 256, 0, stream>>>(wo, wo_bf);

  // QKV projections (pure load+WMMA main loops)
  gemm_nt<<<dim3(M / 64, NQ / 128), 256, 0, stream>>>(x_bf, wq_bf, q_raw, M, NQ, HIDc);
  gemm_nt<<<dim3(M / 64, NK / 128), 256, 0, stream>>>(x_bf, wk_bf, k_raw, M, NK, HIDc);
  gemm_nt<<<dim3(M / 64, NK / 128), 256, 0, stream>>>(x_bf, wv_bf, v_raw, M, NK, HIDc);

  // QK-norm + RoPE (softmax scale folded into Q)
  qknorm_rope<<<Bc * Hc * Sc / 8, 256, 0, stream>>>(
      q_raw, qnw, cosT, sinT, q_bf, Hc, SCALINGc);
  qknorm_rope<<<Bc * KVc * Sc / 8, 256, 0, stream>>>(
      k_raw, knw, cosT, sinT, k_bf, KVc, 1.0f);

  // V -> dim-major bf16
  v_transpose<<<dim3(Sc / 256, Bc * KVc * HDc), 256, 0, stream>>>(v_raw, v_t);

  // Attention
  attn<<<(Bc * Hc * (Sc / 16)) / 8, 256, 0, stream>>>(q_bf, k_bf, v_t, o_bf);

  // Output projection -> f32 result
  gemm_nt<<<dim3(M / 64, HIDc / 128), 256, 0, stream>>>(
      o_bf, wo_bf, out, M, HIDc, NQ);
}
